// DynamicRouting_24678882083219
// MI455X (gfx1250) — matrix operand order repaired and checked by reading, not compiled, not run on hardware
//
#include <hip/hip_runtime.h>
#include <hip/hip_bf16.h>
#include <stdint.h>

typedef __bf16 bf16_t;
typedef __bf16 v16bf __attribute__((ext_vector_type(16)));
typedef __bf16 v8bf  __attribute__((ext_vector_type(8)));
typedef float  v8f   __attribute__((ext_vector_type(8)));

namespace {
constexpr int CAPS = 1024;            // C
constexpr int KCAP = 64;              // K (routing capsules)
constexpr int HDIM = 1024;            // H
constexpr int MROW = CAPS * KCAP;     // 65536 GEMM rows
constexpr int NDIM = HDIM;            // 1024 GEMM cols
constexpr int KDIM = HDIM;            // 1024 reduction dim

constexpr int TM = 128;               // workgroup tile M
constexpr int TN = 256;               // workgroup tile N (4x reuse of X)
constexpr int TK = 32;                // bf16 WMMA K
constexpr int LDSROW = 40;            // bf16 elems per LDS row (32 + 8 pad -> 80B stride)
}

// ---------------------------------------------------------------------------
// f32 -> bf16 pre-conversion (vectorized, n must be a multiple of 4)
// ---------------------------------------------------------------------------
__global__ void cvt_f32_to_bf16_kernel(const float* __restrict__ src,
                                       bf16_t* __restrict__ dst, int n4) {
  int i = blockIdx.x * blockDim.x + threadIdx.x;
  if (i >= n4) return;
  float4 v = ((const float4*)src)[i];
  union { bf16_t b[4]; uint2 u; } o;
  o.b[0] = (bf16_t)v.x;
  o.b[1] = (bf16_t)v.y;
  o.b[2] = (bf16_t)v.z;
  o.b[3] = (bf16_t)v.w;
  ((uint2*)dst)[i] = o.u;
}

// ---------------------------------------------------------------------------
// u_hat[m,n] = sum_h X[m,h] * W[n,h]   (NT GEMM, bf16 in, f32 acc, bf16 out)
// grid = (NDIM/TN, MROW/TM), block = 256
// 8 waves arranged 2 (M) x 4 (N); each wave computes a 64x64 tile
// = 4x4 WMMA fragments -> 16 v_wmma per k-step per wave, 16 ds_load_b128.
// ---------------------------------------------------------------------------
__global__ __launch_bounds__(256) void gemm_uhat_kernel(
    const bf16_t* __restrict__ X, const bf16_t* __restrict__ W,
    bf16_t* __restrict__ U) {
  __shared__ bf16_t As[TM * LDSROW];   // 10 KB
  __shared__ bf16_t Bs[TN * LDSROW];   // 20 KB

  const int tid  = threadIdx.x;
  const int lane = tid & 31;
  const int wave = tid >> 5;
  const int wm   = wave & 1;            // 64-row slab within tile
  const int wn   = wave >> 1;           // 64-col slab within tile

  // Cooperative tile loader: thread -> (row, 16-element K half)
  const int lrow  = tid >> 1;           // 0..127
  const int lhalf = (tid & 1) << 4;     // 0 or 16 (k offset, elements)

  const long gmBase = (long)blockIdx.y * TM;
  const long gnBase = (long)blockIdx.x * TN;

  const bf16_t* aSrc  = X + (gmBase + lrow) * (long)KDIM + lhalf;
  const bf16_t* bSrc0 = W + (gnBase + lrow) * (long)KDIM + lhalf;
  const bf16_t* bSrc1 = W + (gnBase + 128 + lrow) * (long)KDIM + lhalf;
  bf16_t* aDst  = &As[lrow * LDSROW + lhalf];
  bf16_t* bDst0 = &Bs[lrow * LDSROW + lhalf];
  bf16_t* bDst1 = &Bs[(128 + lrow) * LDSROW + lhalf];

  v8f acc[4][4] = {};

  const int laneLo = lane & 15;         // M (A) / N (B) index within fragment
  const int kSel   = lane >> 4;         // which K half-group this lane holds

  for (int kt = 0; kt < KDIM; kt += TK) {
    // ---- stage 128x32 (A) and 256x32 (B) bf16 tiles into LDS ----
    *(uint4*)aDst        = *(const uint4*)(aSrc);
    *(uint4*)(aDst + 8)  = *(const uint4*)(aSrc + 8);
    *(uint4*)bDst0       = *(const uint4*)(bSrc0);
    *(uint4*)(bDst0 + 8) = *(const uint4*)(bSrc0 + 8);
    *(uint4*)bDst1       = *(const uint4*)(bSrc1);
    *(uint4*)(bDst1 + 8) = *(const uint4*)(bSrc1 + 8);
    aSrc  += TK;
    bSrc0 += TK;
    bSrc1 += TK;
    // prefetch two k-steps ahead, WGP-scope (pull into all cache levels)
    __builtin_prefetch(aSrc + TK, 0, 3);
    __builtin_prefetch(bSrc0 + TK, 0, 3);
    __builtin_prefetch(bSrc1 + TK, 0, 3);
    __syncthreads();

    // ---- build WMMA fragments from LDS ----
    union Frag { v16bf v; v8bf h[2]; };
    Frag a[4], b[4];
#pragma unroll
    for (int i = 0; i < 4; ++i) {
      // A 16x32 bf16 layout (ISA 7.12.2): lanes 0-15 hold K {0..7,16..23},
      // lanes 16-31 hold K {8..15,24..31}, row M = lane&15.
      const bf16_t* p = &As[(wm * 64 + i * 16 + laneLo) * LDSROW + kSel * 8];
      a[i].h[0] = *(const v8bf*)(p);
      a[i].h[1] = *(const v8bf*)(p + 16);
    }
#pragma unroll
    for (int j = 0; j < 4; ++j) {
      // B 32x16 bf16 layout: lanes 0-15 hold K {0..7,16..23}, lanes 16-31 hold
      // K {8..15,24..31}, col N = lane&15.
      const bf16_t* p = &Bs[(wn * 64 + j * 16 + laneLo) * LDSROW + kSel * 8];
      b[j].h[0] = *(const v8bf*)(p);
      b[j].h[1] = *(const v8bf*)(p + 16);
    }

    // ---- 16 x v_wmma_f32_16x16x32_bf16 per wave per k-step ----
#pragma unroll
    for (int i = 0; i < 4; ++i)
#pragma unroll
      for (int j = 0; j < 4; ++j)
        acc[i][j] = __builtin_amdgcn_wmma_f32_16x16x32_bf16(
            false, a[i].v, false, b[j].v, (short)0, acc[i][j], false, false);

    __syncthreads();
  }

  // ---- epilogue: C/D layout (lanes 0-15: M=r, lanes 16-31: M=r+8) ----
  const int rowOfs = kSel * 8;
#pragma unroll
  for (int i = 0; i < 4; ++i) {
#pragma unroll
    for (int j = 0; j < 4; ++j) {
      const long r0 = gmBase + wm * 64 + i * 16 + rowOfs;
      const long c0 = gnBase + wn * 64 + j * 16 + laneLo;
#pragma unroll
      for (int r = 0; r < 8; ++r)
        U[(r0 + r) * (long)NDIM + c0] = (bf16_t)acc[i][j][r];
    }
  }
}

// ---------------------------------------------------------------------------
// Fused dynamic routing: one workgroup per capsule c.
// Each of 256 threads keeps u_hat[c, 0..63, h0..h0+3] resident in registers
// (64 x uint2 of packed bf16) -> u_hat read from global exactly once.
// grid = CAPS, block = 256
// ---------------------------------------------------------------------------
__device__ __forceinline__ float bf_lo(unsigned v) {
  union { float f; unsigned u; } t; t.u = v << 16; return t.f;
}
__device__ __forceinline__ float bf_hi(unsigned v) {
  union { float f; unsigned u; } t; t.u = v & 0xFFFF0000u; return t.f;
}

__global__ __launch_bounds__(256) void routing_kernel(
    const bf16_t* __restrict__ U, float* __restrict__ out) {
  __shared__ float b_sh[KCAP];
  __shared__ float d_sh[KCAP];
  __shared__ float red[256];

  const int tid = threadIdx.x;
  const long cap = blockIdx.x;
  const bf16_t* base = U + cap * (long)(KCAP * HDIM) + tid * 4;

  // Register-resident slice: u[k] = 4 bf16 at (k, h0..h0+3)
  uint2 u[KCAP];
#pragma unroll
  for (int k = 0; k < KCAP; ++k)
    u[k] = *(const uint2*)(base + k * HDIM);

  if (tid < KCAP) b_sh[tid] = 0.0f;
  __syncthreads();

  float c0 = 0.f, c1 = 0.f, c2 = 0.f, c3 = 0.f;

  for (int it = 0; it < 3; ++it) {
    // ---- softmax over K=64 (threads 0..63 compute d) ----
    if (tid < KCAP) {
      float m = -1e30f;
      for (int k = 0; k < KCAP; ++k) m = fmaxf(m, b_sh[k]);
      float s = 0.0f;
      for (int k = 0; k < KCAP; ++k) s += __expf(b_sh[k] - m);
      d_sh[tid] = __expf(b_sh[tid] - m) / s;
    }
    __syncthreads();

    // ---- c_hat[h] = sum_k d[k] * u[k,h]  (fully in registers) ----
    float a0 = 0.f, a1 = 0.f, a2 = 0.f, a3 = 0.f;
#pragma unroll
    for (int k = 0; k < KCAP; ++k) {
      const float dk = d_sh[k];
      a0 = fmaf(bf_lo(u[k].x), dk, a0);
      a1 = fmaf(bf_hi(u[k].x), dk, a1);
      a2 = fmaf(bf_lo(u[k].y), dk, a2);
      a3 = fmaf(bf_hi(u[k].y), dk, a3);
    }

    // ---- squash: norm = sum_h c_hat^2 (block tree reduction) ----
    red[tid] = a0 * a0 + a1 * a1 + a2 * a2 + a3 * a3;
    __syncthreads();
    for (int s = 128; s > 0; s >>= 1) {
      if (tid < s) red[tid] += red[tid + s];
      __syncthreads();
    }
    const float norm = red[0];
    const float scale = norm / ((1.0f + norm) * __fsqrt_rn(norm));
    c0 = a0 * scale; c1 = a1 * scale; c2 = a2 * scale; c3 = a3 * scale;

    if (it == 2) break;  // last iteration: b update is unused by reference

    // ---- b[k] += sum_h u[k,h] * c[h]  (wave32 reduce + LDS atomic) ----
#pragma unroll
    for (int k = 0; k < KCAP; ++k) {
      float p = bf_lo(u[k].x) * c0 + bf_hi(u[k].x) * c1 +
                bf_lo(u[k].y) * c2 + bf_hi(u[k].y) * c3;
      p += __shfl_down(p, 16);
      p += __shfl_down(p, 8);
      p += __shfl_down(p, 4);
      p += __shfl_down(p, 2);
      p += __shfl_down(p, 1);
      if ((tid & 31) == 0) atomicAdd(&b_sh[k], p);
    }
    __syncthreads();
  }

  // ---- output c [C,H] in f32 ----
  float4 o;
  o.x = c0; o.y = c1; o.z = c2; o.w = c3;
  *(float4*)(out + cap * (long)HDIM + tid * 4) = o;
}

// ---------------------------------------------------------------------------
// Host launcher
// ---------------------------------------------------------------------------
extern "C" void kernel_launch(void* const* d_in, const int* in_sizes, int n_in,
                              void* d_out, int out_size, void* d_ws, size_t ws_size,
                              hipStream_t stream) {
  (void)in_sizes; (void)n_in; (void)out_size; (void)ws_size;

  const float* X = (const float*)d_in[0];  // encoder_output [C,K,H]
  const float* W = (const float*)d_in[1];  // W [H,H]
  float* out = (float*)d_out;              // c [C,H]

  char* ws = (char*)d_ws;
  const size_t uhatBytes = (size_t)MROW * NDIM * sizeof(bf16_t);  // 128 MB
  const size_t xbfBytes  = (size_t)MROW * KDIM * sizeof(bf16_t);  // 128 MB
  bf16_t* U   = (bf16_t*)(ws);
  bf16_t* Xbf = (bf16_t*)(ws + uhatBytes);
  bf16_t* Wbf = (bf16_t*)(ws + uhatBytes + xbfBytes);             // 2 MB

  // 1) bf16 pre-conversion of X and W
  {
    const int n4x = (MROW * KDIM) / 4;   // 16,777,216
    cvt_f32_to_bf16_kernel<<<n4x / 256, 256, 0, stream>>>(X, Xbf, n4x);
    const int n4w = (HDIM * HDIM) / 4;   // 262,144
    cvt_f32_to_bf16_kernel<<<n4w / 256, 256, 0, stream>>>(W, Wbf, n4w);
  }

  // 2) WMMA GEMM: u_hat (bf16, L2-resident at 134 MB)
  {
    dim3 grid(NDIM / TN, MROW / TM);     // (4, 512)
    gemm_uhat_kernel<<<grid, 256, 0, stream>>>(Xbf, Wbf, U);
  }

  // 3) fused 3-iteration routing, one workgroup per capsule
  routing_kernel<<<CAPS, 256, 0, stream>>>(U, out);
}
